// GlobalPointer_40810779247288
// MI455X (gfx1250) — matrix operand order repaired and checked by reading, not compiled, not run on hardware
//
#include <hip/hip_runtime.h>
#include <hip/hip_bf16.h>
#include <math.h>

// ---------------------------------------------------------------------------
// MI455X (gfx1250) fused RoPE-attention-logits.
//   Stage 0: f32 -> f16 conversion of X (8192x768) and W (1536x768)
//   Stage 1: S = X @ W^T + b with v_wmma_f32_16x16x32_f16, fused RoPE epilogue,
//            writes Q/K f16 tensors [b][h][pos][64] (25 MB total, lives in L2)
//   Stage 2: logits = Q K^T per (b,h) with WMMA, fused mask/causal/scale,
//            402 MB fp32 output (the HBM-bound part: >=17us at 23.3 TB/s).
// All matrix math via WMMA so compute (~32 GFLOP) stays under the HBM floor.
// ---------------------------------------------------------------------------

typedef __attribute__((ext_vector_type(16))) _Float16 v16h;
typedef __attribute__((ext_vector_type(8)))  _Float16 v8h;
typedef __attribute__((ext_vector_type(4)))  _Float16 v4h;
typedef __attribute__((ext_vector_type(8)))  float    v8f;

#define HEADS     12
#define HEAD_SIZE 64
#define HIDDEN    768
#define BATCH     8
#define SEQ       1024
#define QKDIM     (2 * HEADS * HEAD_SIZE)   // 1536
#define MTOT      (BATCH * SEQ)             // 8192

static __device__ __forceinline__ v16h cat8(v8h lo, v8h hi) {
  return __builtin_shufflevector(lo, hi, 0,1,2,3,4,5,6,7,8,9,10,11,12,13,14,15);
}

// A fragment (16x32 f16, row-major source, ld = lda halfwords).
// Lane l (0..15): row mBase+l, K = {0..7, 16..23}; lanes 16..31: same rows? no:
// per ISA: lanes 0-15 M=0..15 hold K{0..7,16..23}; lanes 16-31 M=0..15 hold
// K{8..15,24..31}.  Per lane: two contiguous 8-halfword (b128) loads.
static __device__ __forceinline__ v16h
load_fragA(const _Float16* __restrict__ A, int lda, int mBase, int kBase, int lane) {
  const int row = mBase + (lane & 15);
  const int h8  = ((lane >> 4) & 1) * 8;
  const _Float16* p = A + (size_t)row * lda + kBase + h8;
  v8h c0 = *(const v8h*)(p);
  v8h c1 = *(const v8h*)(p + 16);
  return cat8(c0, c1);
}

// B fragment (32x16 f16). Source is row-major [N x K] (we use it transposed:
// B[k][n] = src[n*ldb + k]).  Lane l: column nBase+(l&15),
// K = kBase + ((l>=16)?8:0) + {0..7, 16..23}: 2x b128 (same order as A).
static __device__ __forceinline__ v16h
load_fragB(const _Float16* __restrict__ Bs, int ldb, int nBase, int kBase, int lane) {
  const int col = nBase + (lane & 15);
  const int k0  = kBase + ((lane >> 4) & 1) * 8;
  const _Float16* p = Bs + (size_t)col * ldb + k0;
  v8h c0 = *(const v8h*)(p);
  v8h c1 = *(const v8h*)(p + 16);
  return cat8(c0, c1);
}

// ---------------- Stage 0: f32 -> f16 conversion ----------------------------
__global__ void cvt_f32_f16(const float* __restrict__ x, _Float16* __restrict__ y, int n4) {
  int i = blockIdx.x * blockDim.x + threadIdx.x;
  if (i < n4) {
    float4 v = ((const float4*)x)[i];
    v4h o = { (_Float16)v.x, (_Float16)v.y, (_Float16)v.z, (_Float16)v.w };
    ((v4h*)y)[i] = o;
  }
}

// ---------------- Stage 1: projection GEMM + RoPE ---------------------------
// Block = 256 threads = 8 waves; block tile 64(M) x 64(N).
// Wave (wave&3) picks M sub-tile, (wave>>2) picks 32-wide N strip (2 accums).
__global__ __launch_bounds__(256) void qkv_rope_kernel(
    const _Float16* __restrict__ Xh,   // [8192 x 768]
    const _Float16* __restrict__ Wh,   // [1536 x 768]
    const float*    __restrict__ bias, // [1536]
    _Float16* __restrict__ Qh,         // [B][H][SEQ][64]
    _Float16* __restrict__ Kh)         // [B][H][SEQ][64]
{
  const int lane  = threadIdx.x & 31;
  const int wave  = threadIdx.x >> 5;
  const int mTile = blockIdx.y * 64 + (wave & 3) * 16;
  const int nBase = blockIdx.x * 64 + (wave >> 2) * 32;

  v8f acc0 = {};
  v8f acc1 = {};
  for (int kk = 0; kk < HIDDEN; kk += 32) {
    v16h a  = load_fragA(Xh, HIDDEN, mTile, kk, lane);
    v16h b0 = load_fragB(Wh, HIDDEN, nBase,      kk, lane);
    v16h b1 = load_fragB(Wh, HIDDEN, nBase + 16, kk, lane);
    acc0 = __builtin_amdgcn_wmma_f32_16x16x32_f16(false, a, false, b0, (short)0, acc0, false, false);
    acc1 = __builtin_amdgcn_wmma_f32_16x16x32_f16(false, a, false, b1, (short)0, acc1, false, false);
  }

  const int hi = (lane >> 4) & 1;
  const int ln = lane & 15;

#pragma unroll
  for (int t = 0; t < 2; ++t) {
    v8f acc;
    if (t == 0) acc = acc0; else acc = acc1;
    const int jcol   = nBase + t * 16 + ln;      // 0..1535
    const int h      = jcol >> 7;                // head
    const int within = jcol & 127;               // 0..127 (q: <64, k: >=64)
    const int d      = jcol & 63;                // dim within q/k head
    const float bj   = bias[jcol];
    // inv_freq = 10000^(-(2*(d>>1))/64) = 2^(-log2(1e4) * (d&~1)/64)
    const float freq = exp2f(-13.287712379549449f * (float)(d & ~1) * (1.0f / 64.0f));
    _Float16* __restrict__ dst = (within < HEAD_SIZE) ? Qh : Kh;

#pragma unroll
    for (int r = 0; r < 8; ++r) {
      const int m   = mTile + r + hi * 8;        // global token index
      const int bz  = m >> 10;
      const int pos = m & (SEQ - 1);
      float x  = acc[r] + bj;
      // RoPE pair partner: column jcol^1 is held by lane^1 (same VGPR r).
      float xp = __shfl_xor(x, 1, 32);
      float s, c;
      __sincosf((float)pos * freq, &s, &c);
      float y = (d & 1) ? fmaf(x, c,  xp * s)    // odd  dim:  x*cos + x_even*sin
                        : fmaf(x, c, -xp * s);   // even dim:  x*cos - x_odd *sin
      dst[((((size_t)bz * HEADS + h) * SEQ) + pos) * HEAD_SIZE + d] = (_Float16)y;
    }
  }
}

// ---------------- Stage 2: attention logits Q K^T ---------------------------
// grid (SEQ/64, SEQ/64, B*H); block tile 64x64, 8 waves, wave tile 16x32.
__global__ __launch_bounds__(256) void attn_logits_kernel(
    const _Float16* __restrict__ Qh,
    const _Float16* __restrict__ Kh,
    const int*      __restrict__ mask,   // [B][SEQ]
    float*          __restrict__ out)    // [B][H][SEQ][SEQ]
{
  const int lane  = threadIdx.x & 31;
  const int wave  = threadIdx.x >> 5;
  const int bh    = blockIdx.z;          // 0..95
  const int bz    = bh / HEADS;
  const int mTile = blockIdx.y * 64 + (wave & 3) * 16;
  const int nBase = blockIdx.x * 64 + (wave >> 2) * 32;

  const _Float16* __restrict__ Qb = Qh + (size_t)bh * SEQ * HEAD_SIZE;
  const _Float16* __restrict__ Kb = Kh + (size_t)bh * SEQ * HEAD_SIZE;

  v8f acc0 = {};
  v8f acc1 = {};
#pragma unroll
  for (int kk = 0; kk < HEAD_SIZE; kk += 32) {
    v16h a  = load_fragA(Qb, HEAD_SIZE, mTile, kk, lane);
    v16h b0 = load_fragB(Kb, HEAD_SIZE, nBase,      kk, lane);  // B[k][n] = K[n][k]
    v16h b1 = load_fragB(Kb, HEAD_SIZE, nBase + 16, kk, lane);
    acc0 = __builtin_amdgcn_wmma_f32_16x16x32_f16(false, a, false, b0, (short)0, acc0, false, false);
    acc1 = __builtin_amdgcn_wmma_f32_16x16x32_f16(false, a, false, b1, (short)0, acc1, false, false);
  }

  const int hi = (lane >> 4) & 1;
  const int ln = lane & 15;
  const int* __restrict__ mb = mask + bz * SEQ;
  float* __restrict__ ob = out + (size_t)bh * SEQ * SEQ;

#pragma unroll
  for (int t = 0; t < 2; ++t) {
    v8f acc;
    if (t == 0) acc = acc0; else acc = acc1;
    const int n   = nBase + t * 16 + ln;
    const bool vc = mb[n] != 0;
#pragma unroll
    for (int r = 0; r < 8; ++r) {
      const int m = mTile + r + hi * 8;
      float v = acc[r];
      if (!vc || mb[m] == 0) v = -__builtin_inff();       // invalid -> -inf
      if (m > n)            v -= 1.0e12f;                  // strict causal penalty
      v *= 0.125f;                                         // 1/sqrt(64)
      ob[(size_t)m * SEQ + n] = v;
    }
  }
}

// ---------------------------------------------------------------------------
extern "C" void kernel_launch(void* const* d_in, const int* in_sizes, int n_in,
                              void* d_out, int out_size, void* d_ws, size_t ws_size,
                              hipStream_t stream) {
  const float* inputs = (const float*)d_in[0];   // (8,1024,768) f32
  const int*   mask   = (const int*)  d_in[1];   // (8,1024) i32
  const float* W      = (const float*)d_in[2];   // (1536,768) f32
  const float* bias   = (const float*)d_in[3];   // (1536,) f32
  float*       out    = (float*)d_out;           // (8,12,1024,1024) f32

  char* ws = (char*)d_ws;
  size_t o = 0;
  _Float16* Xh = (_Float16*)(ws + o); o += (size_t)MTOT * HIDDEN * sizeof(_Float16);
  _Float16* Wh = (_Float16*)(ws + o); o += (size_t)QKDIM * HIDDEN * sizeof(_Float16);
  _Float16* Qh = (_Float16*)(ws + o); o += (size_t)BATCH * HEADS * SEQ * HEAD_SIZE * sizeof(_Float16);
  _Float16* Kh = (_Float16*)(ws + o);

  // Stage 0: conversions (vectorized, counts divisible by 4)
  {
    int n4 = (MTOT * HIDDEN) / 4;          // 1,572,864
    cvt_f32_f16<<<(n4 + 255) / 256, 256, 0, stream>>>(inputs, Xh, n4);
    int w4 = (QKDIM * HIDDEN) / 4;         // 294,912
    cvt_f32_f16<<<(w4 + 255) / 256, 256, 0, stream>>>(W, Wh, w4);
  }

  // Stage 1: projection + RoPE.  grid: (1536/64, 8192/64)
  {
    dim3 g(QKDIM / 64, MTOT / 64);
    qkv_rope_kernel<<<g, 256, 0, stream>>>(Xh, Wh, bias, Qh, Kh);
  }

  // Stage 2: logits.  grid: (1024/64, 1024/64, 96)
  {
    dim3 g(SEQ / 64, SEQ / 64, BATCH * HEADS);
    attn_logits_kernel<<<g, 256, 0, stream>>>(Qh, Kh, mask, out);
  }
}
